// FusedMoEModularKernel_37812892074042
// MI455X (gfx1250) — matrix operand order repaired and checked by reading, not compiled, not run on hardware
//
#include <hip/hip_runtime.h>
#include <hip/hip_bf16.h>

// Problem constants (match reference)
#define M_TOK 4096
#define K_DIM 2048
#define N2_DIM 2048
#define N_DIM 1024
#define E_NUM 16
#define TOPK 2
#define CAP (M_TOK * TOPK)        // worst-case tokens per expert
#define TILE_M 32                 // tokens per block (2 WMMA m-tiles)
#define MAX_TILES (CAP / TILE_M)  // 256

// LDS row padding (in bf16 elems) to de-conflict banks: rows 4 banks apart
#define XPAD 8
#define XSTR (K_DIM + XPAD)       // 2056
#define ASTR (N_DIM + XPAD)       // 1032

typedef __attribute__((ext_vector_type(16))) __bf16 v16bf;
typedef __attribute__((ext_vector_type(8)))  float  v8f;

static __device__ __forceinline__ v8f wmma_bf16(v16bf a, v16bf b, v8f c) {
  // 8 args: (neg_a, A, neg_b, B, c_mod, C, reuse_a, reuse_b)
  return __builtin_amdgcn_wmma_f32_16x16x32_bf16(false, a, false, b, (short)0, c,
                                                 false, false);
}

// B fragment (32x16, 16-bit): lane l holds column n = l%16;
// elems 0..7 -> K = (l/16)*8 + i ; elems 8..15 -> K = 16 + (l/16)*8 + i.
// p points at &row[kb + (l/16)*8] of the fp32 weight row -> convert to bf16.
static __device__ __forceinline__ v16bf load16_cvt(const float* __restrict__ p) {
  v16bf r;
#pragma unroll
  for (int i = 0; i < 8; ++i) { r[i] = (__bf16)p[i]; r[8 + i] = (__bf16)p[16 + i]; }
  return r;
}

// A fragment (16x32, 16-bit): lane l holds row m = l%16;
// elems 0..7  -> K = kb + (l/16)*8 + i ; elems 8..15 -> K = kb + 16 + (l/16)*8 + i
// rowp points at &row[kb + (l/16)*8]
static __device__ __forceinline__ v16bf loadA_lds(const __bf16* __restrict__ rowp) {
  v16bf r;
#pragma unroll
  for (int i = 0; i < 8; ++i) { r[i] = rowp[i]; r[8 + i] = rowp[16 + i]; }
  return r;
}

__global__ void zero_kernel(float* __restrict__ out, int* __restrict__ counts) {
  int idx = blockIdx.x * blockDim.x + threadIdx.x;
  float4 z = {0.f, 0.f, 0.f, 0.f};
  if (idx < (M_TOK * K_DIM) / 4) ((float4*)out)[idx] = z;
  if (idx < E_NUM) counts[idx] = 0;
}

__global__ void route_kernel(const float* __restrict__ tw,
                             const int* __restrict__ ids,
                             int* __restrict__ counts,
                             int* __restrict__ tokList,
                             float* __restrict__ wgtList) {
  int i = blockIdx.x * blockDim.x + threadIdx.x;
  if (i >= M_TOK * TOPK) return;
  int m = i / TOPK;
  int e = ids[i];
  int slot = atomicAdd(&counts[e], 1);
  tokList[e * CAP + slot] = m;
  wgtList[e * CAP + slot] = tw[i];
}

__global__ __launch_bounds__(256, 1)
void moe_kernel(const float* __restrict__ a1,
                const float* __restrict__ w1,
                const float* __restrict__ w2,
                float* __restrict__ out,
                const int* __restrict__ counts,
                const int* __restrict__ tokList,
                const float* __restrict__ wgtList) {
  __shared__ __bf16 X[TILE_M][XSTR];    // ~128.5 KB gathered activations (bf16)
  __shared__ __bf16 Act[TILE_M][ASTR];  // ~64.5 KB silu(gate)*up (bf16)
  __shared__ int   tok[TILE_M];
  __shared__ float wgt[TILE_M];

  const int e    = blockIdx.y;
  const int tIdx = blockIdx.x;
  const int cnt  = counts[e];
  if (tIdx * TILE_M >= cnt) return;

  const float* __restrict__ w1e = w1 + (size_t)e * N2_DIM * K_DIM;
  const float* __restrict__ w2e = w2 + (size_t)e * K_DIM * N_DIM;

  const int tid = threadIdx.x;
  if (tid < TILE_M) {
    int idx = tIdx * TILE_M + tid;
    if (idx < cnt) {
      tok[tid] = tokList[e * CAP + idx];
      wgt[tid] = wgtList[e * CAP + idx];
    } else {  // pad with a valid row, zero routing weight -> contributes 0
      tok[tid] = tokList[e * CAP + tIdx * TILE_M];
      wgt[tid] = 0.0f;
    }
  }
  __syncthreads();

  // ---- Gather X tile: TILE_M rows x K fp32 -> bf16 LDS (float4 loads) ----
  for (int c = tid; c < TILE_M * (K_DIM / 4); c += 256) {
    int r  = c / (K_DIM / 4);
    int k4 = c % (K_DIM / 4);
    float4 v = ((const float4*)(a1 + (size_t)tok[r] * K_DIM))[k4];
    __bf16* xp = &X[r][k4 * 4];
    xp[0] = (__bf16)v.x; xp[1] = (__bf16)v.y;
    xp[2] = (__bf16)v.z; xp[3] = (__bf16)v.w;
  }
  __syncthreads();

  const int wave = tid >> 5;
  const int lane = tid & 31;
  const int half = lane >> 4;    // 0: lanes 0-15, 1: lanes 16-31
  const int l16  = lane & 15;

  // ---- Phase A: h = X @ w1e^T ; act = silu(gate)*up -> bf16 LDS ----
  // 64 n-tiles of 16 cols; each wave owns 8 of them; both m-tiles per iter
  // share the gate/up B fragments (2 WMMAs per converted fragment).
  for (int nt = wave; nt < N_DIM / 16; nt += 8) {
    v8f aG0 = {}, aU0 = {}, aG1 = {}, aU1 = {};
    const float* __restrict__ wgrow = w1e + (size_t)(nt * 16 + l16) * K_DIM;
    const float* __restrict__ wurow = w1e + (size_t)(N_DIM + nt * 16 + l16) * K_DIM;
    if (nt + 8 < N_DIM / 16) {  // prefetch next owned n-tile's rows
      __builtin_prefetch(w1e + (size_t)((nt + 8) * 16 + l16) * K_DIM, 0, 1);
      __builtin_prefetch(w1e + (size_t)(N_DIM + (nt + 8) * 16 + l16) * K_DIM, 0, 1);
    }
#pragma unroll 2
    for (int kt = 0; kt < K_DIM / 32; ++kt) {
      int kb = kt * 32;
      v16bf a0 = loadA_lds(&X[l16][kb + half * 8]);
      v16bf a1f = loadA_lds(&X[16 + l16][kb + half * 8]);
      v16bf bg = load16_cvt(wgrow + kb + half * 8);
      v16bf bu = load16_cvt(wurow + kb + half * 8);
      aG0 = wmma_bf16(a0,  bg, aG0);
      aG1 = wmma_bf16(a1f, bg, aG1);
      aU0 = wmma_bf16(a0,  bu, aU0);
      aU1 = wmma_bf16(a1f, bu, aU1);
    }
#pragma unroll
    for (int r = 0; r < 8; ++r) {
      int row = r + half * 8;  // D layout: M = vgpr + 8*(lane/16)
      float g0 = aG0[r], u0 = aU0[r];
      float g1 = aG1[r], u1 = aU1[r];
      Act[row][nt * 16 + l16]      = (__bf16)((g0 / (1.0f + __expf(-g0))) * u0);
      Act[16 + row][nt * 16 + l16] = (__bf16)((g1 / (1.0f + __expf(-g1))) * u1);
    }
  }
  __syncthreads();

  // ---- Phase B: y = act @ w2e^T ; scale by routing weight; atomic combine ----
  // 128 output k-tiles of 16; each wave owns 16; both m-tiles share B fragment.
  for (int ko = wave; ko < K_DIM / 16; ko += 8) {
    v8f acc0 = {}, acc1 = {};
    const float* __restrict__ wrow = w2e + (size_t)(ko * 16 + l16) * N_DIM;
    if (ko + 8 < K_DIM / 16) {
      __builtin_prefetch(w2e + (size_t)((ko + 8) * 16 + l16) * N_DIM, 0, 1);
    }
#pragma unroll 2
    for (int nt = 0; nt < N_DIM / 32; ++nt) {
      int nb = nt * 32;
      v16bf a0 = loadA_lds(&Act[l16][nb + half * 8]);
      v16bf a1f = loadA_lds(&Act[16 + l16][nb + half * 8]);
      v16bf b = load16_cvt(wrow + nb + half * 8);
      acc0 = wmma_bf16(a0,  b, acc0);
      acc1 = wmma_bf16(a1f, b, acc1);
    }
#pragma unroll
    for (int r = 0; r < 8; ++r) {
      int row = r + half * 8;
      atomicAdd(&out[(size_t)tok[row] * K_DIM + ko * 16 + l16],
                acc0[r] * wgt[row]);
      atomicAdd(&out[(size_t)tok[16 + row] * K_DIM + ko * 16 + l16],
                acc1[r] * wgt[16 + row]);
    }
  }
}

extern "C" void kernel_launch(void* const* d_in, const int* in_sizes, int n_in,
                              void* d_out, int out_size, void* d_ws, size_t ws_size,
                              hipStream_t stream) {
  const float* a1  = (const float*)d_in[0];
  const float* w1  = (const float*)d_in[1];
  const float* w2  = (const float*)d_in[2];
  const float* tw  = (const float*)d_in[3];
  const int*   ids = (const int*)d_in[4];
  float* out = (float*)d_out;

  // Workspace layout
  char* ws = (char*)d_ws;
  int*   counts  = (int*)ws;                          // E ints
  int*   tokList = (int*)(ws + 256);                  // E*CAP ints (512 KB)
  float* wgtList = (float*)(ws + 256 + sizeof(int) * E_NUM * CAP);  // 512 KB

  // 1) zero output + expert counters
  {
    int n4 = (M_TOK * K_DIM) / 4;
    zero_kernel<<<(n4 + 255) / 256, 256, 0, stream>>>(out, counts);
  }
  // 2) routing scatter
  {
    int n = M_TOK * TOPK;
    route_kernel<<<(n + 255) / 256, 256, 0, stream>>>(tw, ids, counts, tokList, wgtList);
  }
  // 3) grouped gated-MLP GEMM (32-token tiles, 8 waves/block)
  {
    dim3 grid(MAX_TILES, E_NUM);
    moe_kernel<<<grid, 256, 0, stream>>>(a1, w1, w2, out, counts, tokList, wgtList);
  }
}
